// StarTransformer_50783693308406
// MI455X (gfx1250) — matrix operand order repaired and checked by reading, not compiled, not run on hardware
//
#include <hip/hip_runtime.h>
#include <hip/hip_bf16.h>
#include <math.h>

typedef __bf16 bf16;
typedef __attribute__((ext_vector_type(16))) __bf16 v16bf;
typedef __attribute__((ext_vector_type(8)))  float  v8f;

#define HDIM   512
#define HEADS  8
#define DK     64
#define DEG    4
#define NLAYER 5
#define GATE3  1536   // [W_z; W_r; W_h] rows
#define MTILES 96     // 1536/16
#define KSTEPS 16     // 512/32

// ---------------------------------------------------------------------------
// 1) Transpose E_bu [H,VOCAB] -> Et [VOCAB,H] (LDS-tiled, coalesced both ways)
// ---------------------------------------------------------------------------
__global__ void k_transpose(const float* __restrict__ E, float* __restrict__ Et, int vocab) {
    __shared__ float tile[16][17];
    int tx = threadIdx.x, ty = threadIdx.y;
    int v = blockIdx.x * 16 + tx;
    int h = blockIdx.y * 16 + ty;
    if (v < vocab) tile[ty][tx] = E[(size_t)h * vocab + v];
    __syncthreads();
    int vo = blockIdx.x * 16 + ty;
    int ho = blockIdx.y * 16 + tx;
    if (vo < vocab) Et[(size_t)vo * HDIM + ho] = tile[tx][ty];
}

// ---------------------------------------------------------------------------
// 2) Embedding: xe[n] = sum_l Et[idx[n,l]] * w[n,l]. Leaves -> node_h, parents -> xe_par
// ---------------------------------------------------------------------------
__global__ void k_embed(const float* __restrict__ xw, const int* __restrict__ xi,
                        const float* __restrict__ Et, float* __restrict__ node_h,
                        float* __restrict__ xe_par, int L) {
    __shared__ float sw[32];
    __shared__ int   si[32];
    int n = blockIdx.x;
    int t = threadIdx.x;  // 256 threads, each owns dims t and t+256
    if (t < 32) { sw[t] = xw[(size_t)n * 32 + t]; si[t] = xi[(size_t)n * 32 + t]; }
    __syncthreads();
    float a0 = 0.f, a1 = 0.f;
    for (int l = 0; l < 32; ++l) {
        const float* row = Et + (size_t)si[l] * HDIM;
        float w = sw[l];
        a0 += row[t] * w;
        a1 += row[t + 256] * w;
    }
    if (n < L) {
        node_h[(size_t)n * HDIM + t] = a0;
        node_h[(size_t)n * HDIM + t + 256] = a1;
    } else {
        xe_par[(size_t)(n - L) * HDIM + t] = a0;
        xe_par[(size_t)(n - L) * HDIM + t + 256] = a1;
    }
}

// ---------------------------------------------------------------------------
// 3) Pack A = [W_z;W_r;W_h] (1536x512 f32) into wave32 WMMA bf16 A-fragment layout.
//    A 16x32 bf16 layout (ISA 7.12.2): lane = hl*16+m; vgpr j pair t ->
//    K = (j<4) ? hl*8+2j+t : 16+hl*8+2(j-4)+t
// ---------------------------------------------------------------------------
__global__ void k_packA(const float* __restrict__ Wz, const float* __restrict__ Wr,
                        const float* __restrict__ Wh, bf16* __restrict__ Apack) {
    int idx = blockIdx.x * 256 + threadIdx.x;        // total 96*16*32*16 = 786432
    int e    = idx & 15;
    int lane = (idx >> 4) & 31;
    int ks   = (idx >> 9) & 15;
    int mt   = idx >> 13;
    int hl = lane >> 4, ml = lane & 15;
    int m = mt * 16 + ml;
    int j = e >> 1, tt = e & 1;
    int kl = (j < 4) ? (hl * 8 + 2 * j + tt) : (16 + hl * 8 + 2 * (j - 4) + tt);
    int k = ks * 32 + kl;
    float v;
    if (m < 512)        v = Wz[(size_t)m * HDIM + k];
    else if (m < 1024)  v = Wr[(size_t)(m - 512) * HDIM + k];
    else                v = Wh[(size_t)(m - 1024) * HDIM + k];
    Apack[idx] = (bf16)v;
}

// ---------------------------------------------------------------------------
// 4) Pack B[k,n] = xe_par[n,k] into wave32 WMMA bf16 B-fragment layout.
//    B 32x16 bf16 layout: lane = hl*16+n; vgpr j pair t -> K = (j<4) ? hl*8+2j+t : 16+hl*8+2(j-4)+t
// ---------------------------------------------------------------------------
__global__ void k_packB(const float* __restrict__ xe_par, bf16* __restrict__ Bpack,
                        int P, int Ptiles) {
    int idx = blockIdx.x * 256 + threadIdx.x;
    int total = Ptiles * 16 * 512;
    if (idx >= total) return;
    int e    = idx & 15;
    int lane = (idx >> 4) & 31;
    int ks   = (idx >> 9) & 15;
    int nt   = idx >> 13;
    int hl = lane >> 4, nl = lane & 15;
    int p = nt * 16 + nl;
    int j = e >> 1, tt = e & 1;
    int k = ks * 32 + ((j < 4) ? (hl * 8 + 2 * j + tt) : (16 + hl * 8 + 2 * (j - 4) + tt));
    Bpack[idx] = (p < P) ? (bf16)xe_par[(size_t)p * HDIM + k] : (bf16)0.0f;
}

// ---------------------------------------------------------------------------
// 5) WMMA GEMM: Wx[p, m] = sum_k Wcat[m,k]*xe_par[p,k].  One 16x16 tile per wave.
// ---------------------------------------------------------------------------
__global__ void k_gemm(const bf16* __restrict__ Apack, const bf16* __restrict__ Bpack,
                       float* __restrict__ Wx, int P, int Ptiles) {
    int wid  = (blockIdx.x * blockDim.x + threadIdx.x) >> 5;
    int lane = threadIdx.x & 31;
    if (wid >= MTILES * Ptiles) return;
    int mt = wid % MTILES;
    int nt = wid / MTILES;   // consecutive waves share the B tile -> L2 reuse

    const bf16* abase = Apack + ((size_t)(mt * KSTEPS) * 32 + lane) * 16;
    const bf16* bbase = Bpack + ((size_t)(nt * KSTEPS) * 32 + lane) * 16;

    v8f acc = {};
#pragma unroll
    for (int ks = 0; ks < KSTEPS; ++ks) {
        v16bf a = *(const v16bf*)(abase + (size_t)ks * 512);
        v16bf b = *(const v16bf*)(bbase + (size_t)ks * 512);
        acc = __builtin_amdgcn_wmma_f32_16x16x32_bf16(false, a, false, b,
                                                      (short)0, acc, false, false);
    }

    // D 16x16 f32: lane L -> col n = L&15 (+16 half), VGPR v -> row m = v + 8*(L>>4)
    int hl = lane >> 4, nl = lane & 15;
    int p = nt * 16 + nl;
    if (p < P) {
        float* out = Wx + (size_t)p * GATE3 + mt * 16 + hl * 8;
#pragma unroll
        for (int v = 0; v < 8; ++v) out[v] = acc[v];
    }
}

// ---------------------------------------------------------------------------
// 6) Serial scan: one persistent 512-thread (16-wave) block over P parents.
// ---------------------------------------------------------------------------
__global__ void __launch_bounds__(512)
k_scan(const int* __restrict__ tree, const float* __restrict__ Wx,
       const float* __restrict__ Uz, const float* __restrict__ Ur, const float* __restrict__ Uh,
       const float* __restrict__ bz, const float* __restrict__ br, const float* __restrict__ bh,
       float* __restrict__ node_h, int L, int P) {
    __shared__ float hbuf[2][DEG][HDIM];     // attention ping-pong
    __shared__ float s_mem[HDIM], s_z[HDIM], s_r[HDIM], s_c[HDIM], s_rm[HDIM];
    __shared__ float sc[HEADS * 16], pr[HEADS * 16];
    __shared__ int   s_ch[DEG];

    int t = threadIdx.x;          // 0..511
    int w = t >> 5, lane = t & 31;

    for (int p = 0; p < P; ++p) {
        if (t < DEG) s_ch[t] = tree[(size_t)p * DEG + t];
        __syncthreads();
        int msk[DEG];
#pragma unroll
        for (int j = 0; j < DEG; ++j) msk[j] = (s_ch[j] > -1);

        // gather masked children h
#pragma unroll
        for (int j = 0; j < DEG; ++j) {
            float v = 0.f;
            if (msk[j]) v = node_h[(size_t)s_ch[j] * HDIM + t];
            hbuf[0][j][t] = v;
        }
        __syncthreads();

        int cur = 0;
        for (int layer = 0; layer < NLAYER; ++layer) {
            // scores[head][i][j] = q_i . q_j / sqrt(64), keys masked to -1e9
            if (t < 128) {
                int head = t >> 4, i = (t >> 2) & 3, j = t & 3;
                const float* qi = &hbuf[cur][i][head * DK];
                const float* qj = &hbuf[cur][j][head * DK];
                float s = 0.f;
                for (int d = 0; d < DK; ++d) s += qi[d] * qj[d];
                sc[t] = msk[j] ? s * 0.125f : -1e9f;
            }
            __syncthreads();
            // softmax over j (4 keys)
            if (t < 32) {
                int base = t * 4;
                float m = fmaxf(fmaxf(sc[base], sc[base + 1]), fmaxf(sc[base + 2], sc[base + 3]));
                float e0 = expf(sc[base] - m),     e1 = expf(sc[base + 1] - m);
                float e2 = expf(sc[base + 2] - m), e3 = expf(sc[base + 3] - m);
                float inv = 1.f / (e0 + e1 + e2 + e3);
                pr[base] = e0 * inv; pr[base + 1] = e1 * inv;
                pr[base + 2] = e2 * inv; pr[base + 3] = e3 * inv;
            }
            __syncthreads();
            // out[i][t] = sum_j p[head,i,j] * h[j][t]
            int head = t >> 6;
            float c0 = hbuf[cur][0][t], c1 = hbuf[cur][1][t];
            float c2 = hbuf[cur][2][t], c3 = hbuf[cur][3][t];
#pragma unroll
            for (int i = 0; i < DEG; ++i) {
                const float* pp = &pr[head * 16 + i * 4];
                hbuf[cur ^ 1][i][t] = pp[0] * c0 + pp[1] * c1 + pp[2] * c2 + pp[3] * c3;
            }
            __syncthreads();
            cur ^= 1;
        }

        // memory = masked mean of attended children
        {
            int cnt = msk[0] + msk[1] + msk[2] + msk[3];
            float inv = 1.f / (float)(cnt > 0 ? cnt : 1);
            float s = 0.f;
#pragma unroll
            for (int j = 0; j < DEG; ++j) if (msk[j]) s += hbuf[cur][j][t];
            s_mem[t] = s * inv;
        }
        __syncthreads();

        const float* wxp = Wx + (size_t)p * GATE3;

        // z (rows 0..511) and r (rows 512..1023): 64 rows per wave, lanes stride K
        for (int rr = 0; rr < 64; ++rr) {
            int m = w * 64 + rr;
            const float* Urow = (m < 512) ? (Uz + (size_t)m * HDIM)
                                          : (Ur + (size_t)(m - 512) * HDIM);
            if (rr + 1 < 64) {
                int mn = m + 1;
                const float* nrow = (mn < 512) ? (Uz + (size_t)mn * HDIM)
                                               : (Ur + (size_t)(mn - 512) * HDIM);
                __builtin_prefetch(nrow + lane * 16, 0, 3);  // global_prefetch_b8
            }
            float acc = 0.f;
#pragma unroll
            for (int k = lane; k < HDIM; k += 32) acc += Urow[k] * s_mem[k];
#pragma unroll
            for (int off = 16; off > 0; off >>= 1) acc += __shfl_xor(acc, off, 32);
            if (lane == 0) {
                if (m < 512) {
                    float pre = wxp[m] + acc + bz[m];
                    s_z[m] = 1.f / (1.f + expf(-pre));
                } else {
                    float pre = wxp[m] + acc + br[m - 512];
                    s_r[m - 512] = 1.f / (1.f + expf(-pre));
                }
            }
        }
        __syncthreads();
        s_rm[t] = s_mem[t] * s_r[t];
        __syncthreads();

        // c: U_h @ (mem * r), 32 rows per wave
        for (int rr = 0; rr < 32; ++rr) {
            int m = w * 32 + rr;
            const float* Urow = Uh + (size_t)m * HDIM;
            if (rr + 1 < 32) __builtin_prefetch(Urow + HDIM + lane * 16, 0, 3);
            float acc = 0.f;
#pragma unroll
            for (int k = lane; k < HDIM; k += 32) acc += Urow[k] * s_rm[k];
#pragma unroll
            for (int off = 16; off > 0; off >>= 1) acc += __shfl_xor(acc, off, 32);
            if (lane == 0) {
                float pre = wxp[1024 + m] + acc + bh[m];
                s_c[m] = tanhf(pre);
            }
        }
        __syncthreads();

        float z = s_z[t];
        float hp = z * s_mem[t] + (1.f - z) * s_c[t];
        node_h[(size_t)(L + p) * HDIM + t] = hp;
        __threadfence();   // make parent h visible to next iteration's gathers
        __syncthreads();
    }
}

// ---------------------------------------------------------------------------
// 7) Column-wise max over parents, then softmax(W_out @ final + b_out)
// ---------------------------------------------------------------------------
__global__ void k_final(const float* __restrict__ node_h, const float* __restrict__ Wout,
                        const float* __restrict__ bout, float* __restrict__ out,
                        int L, int P) {
    __shared__ float fin[HDIM];
    __shared__ float lg[4];
    int t = threadIdx.x;
    float m = -3.4e38f;
    const float* base = node_h + (size_t)L * HDIM + t;
    for (int p = 0; p < P; ++p) m = fmaxf(m, base[(size_t)p * HDIM]);
    fin[t] = m;
    __syncthreads();
    int w = t >> 5, lane = t & 31;
    if (w < 4) {
        const float* row = Wout + (size_t)w * HDIM;
        float acc = 0.f;
        for (int k = lane; k < HDIM; k += 32) acc += row[k] * fin[k];
#pragma unroll
        for (int off = 16; off > 0; off >>= 1) acc += __shfl_xor(acc, off, 32);
        if (lane == 0) lg[w] = acc + bout[w];
    }
    __syncthreads();
    if (t == 0) {
        float mm = fmaxf(fmaxf(lg[0], lg[1]), fmaxf(lg[2], lg[3]));
        float e0 = expf(lg[0] - mm), e1 = expf(lg[1] - mm);
        float e2 = expf(lg[2] - mm), e3 = expf(lg[3] - mm);
        float inv = 1.f / (e0 + e1 + e2 + e3);
        out[0] = e0 * inv; out[1] = e1 * inv; out[2] = e2 * inv; out[3] = e3 * inv;
    }
}

// ---------------------------------------------------------------------------
extern "C" void kernel_launch(void* const* d_in, const int* in_sizes, int n_in,
                              void* d_out, int out_size, void* d_ws, size_t ws_size,
                              hipStream_t stream) {
    const float* x_word  = (const float*)d_in[0];
    const int*   x_index = (const int*)  d_in[1];
    const int*   tree    = (const int*)  d_in[2];
    const float* E_bu    = (const float*)d_in[3];
    const float* W_z = (const float*)d_in[4];
    const float* U_z = (const float*)d_in[5];
    const float* b_z = (const float*)d_in[6];
    const float* W_r = (const float*)d_in[7];
    const float* U_r = (const float*)d_in[8];
    const float* b_r = (const float*)d_in[9];
    const float* W_h = (const float*)d_in[10];
    const float* U_h = (const float*)d_in[11];
    const float* b_h = (const float*)d_in[12];
    const float* W_out = (const float*)d_in[13];
    const float* b_out = (const float*)d_in[14];

    int N = in_sizes[0] / 32;
    int P = in_sizes[2] / 4;
    int L = N - P;
    int VOCAB = in_sizes[3] / HDIM;
    int Ptiles = (P + 15) / 16;

    char* ws = (char*)d_ws;
    size_t off = 0;
    auto alloc = [&](size_t bytes) -> char* {
        char* r = ws + off;
        off += (bytes + 255) & ~(size_t)255;
        return r;
    };
    float* Et     = (float*)alloc((size_t)VOCAB * HDIM * 4);
    float* node_h = (float*)alloc((size_t)N * HDIM * 4);
    float* xe_par = (float*)alloc((size_t)P * HDIM * 4);
    float* Wx     = (float*)alloc((size_t)P * GATE3 * 4);
    bf16*  Apack  = (bf16*)alloc((size_t)MTILES * KSTEPS * 512 * 2);
    bf16*  Bpack  = (bf16*)alloc((size_t)Ptiles * KSTEPS * 512 * 2);
    (void)ws_size; (void)n_in; (void)out_size;

    {
        dim3 tb(16, 16);
        dim3 tg((VOCAB + 15) / 16, HDIM / 16);
        k_transpose<<<tg, tb, 0, stream>>>(E_bu, Et, VOCAB);
    }
    k_embed<<<N, 256, 0, stream>>>(x_word, x_index, Et, node_h, xe_par, L);
    k_packA<<<(MTILES * KSTEPS * 512) / 256, 256, 0, stream>>>(W_z, W_r, W_h, Apack);
    {
        int totalB = Ptiles * KSTEPS * 512;
        k_packB<<<(totalB + 255) / 256, 256, 0, stream>>>(xe_par, Bpack, P, Ptiles);
    }
    {
        int waves = MTILES * Ptiles;
        int blocks = (waves * 32 + 255) / 256;
        k_gemm<<<blocks, 256, 0, stream>>>(Apack, Bpack, Wx, P, Ptiles);
    }
    k_scan<<<1, 512, 0, stream>>>(tree, Wx, U_z, U_r, U_h, b_z, b_r, b_h, node_h, L, P);
    k_final<<<1, 512, 0, stream>>>(node_h, W_out, b_out, (float*)d_out, L, P);
}
